// Decoder_9517647528302
// MI455X (gfx1250) — matrix operand order repaired and checked by reading, not compiled, not run on hardware
//
#include <hip/hip_runtime.h>

// ---------------------------------------------------------------------------
// Decoder with Bahdanau attention + coverage + GRU cell, for MI455X (gfx1250).
// B=64, L=128, N=512, E=512, H=512.
//   - One-time: convert weights/value to bf16, pack [Wq;W_hh] -> one GEMM,
//     vk = value @ Wk^T stored bf16 (halves the per-step L2 stream).
//   - Per step: fused (q|gh) WMMA GEMM, fused attention (score+softmax+context
//     over bf16 streams), gi WMMA GEMM, elementwise GRU gate kernel.
//   - GEMM: v_wmma_f32_16x16x32_bf16, BM=64/BN=64/KC=64, 2 accumulators/wave,
//     double-buffered LDS, A tiles staged with global_load_async_to_lds_b128
//     (ASYNCcnt), one barrier per k-stage.
// ---------------------------------------------------------------------------

typedef __attribute__((ext_vector_type(16))) __bf16 v16bf;
typedef __attribute__((ext_vector_type(8)))  float  v8f;
typedef unsigned short ushort_t;

union FragU { uint4 q[2]; v16bf v; };

__device__ __forceinline__ ushort_t f2bf(float f) {
    unsigned u = __float_as_uint(f);
    u += 0x7fffu + ((u >> 16) & 1u);      // round-to-nearest-even
    return (ushort_t)(u >> 16);
}
__device__ __forceinline__ float bf2f(ushort_t u) {
    return __uint_as_float(((unsigned)u) << 16);
}

// gfx1250 async global->LDS copy (VGPR-bypassing, tracked by ASYNCcnt).
__device__ __forceinline__ void async_copy_b128(const void* gptr, void* lptr) {
    unsigned lds_off = (unsigned)(unsigned long long)lptr;  // low 32b = LDS offset
    asm volatile("global_load_async_to_lds_b128 %0, %1, off"
                 :: "v"(lds_off), "v"(gptr) : "memory");
}
__device__ __forceinline__ void wait_async0() {
    asm volatile("s_wait_asynccnt 0x0" ::: "memory");
}

// ---------------------------------------------------------------------------
// NT GEMM, bf16 inputs: C[m,n] = sum_k A[m,k]*Bm[n,k] (+bias[n]).
// Tile BM=64 x BN=64 x KC=64; 8 waves (2x4); each wave owns two 16x16 tiles
// (m and m+32) -> 4 WMMAs per k-stage with B-fragment reuse. Double-buffered
// LDS, async A staging, one barrier per stage.
// Requires M%64==0, N%64==0, K%64==0 (true for all call sites).
// ---------------------------------------------------------------------------
#define GBM 64
#define GBN 64
#define GKC 64
#define SA_S 72   // ushort row stride (144B, 16B-aligned rows)
#define SB_S 72

template <bool OUT_BF16, bool HAS_BIAS>
__global__ __launch_bounds__(256)
void gemm_nt_bf16(const ushort_t* __restrict__ A, int lda,
                  const ushort_t* __restrict__ Bm, int ldb,
                  void* __restrict__ Cv, int ldc,
                  const float* __restrict__ bias, int K) {
    __shared__ __align__(16) ushort_t sA[2][GBM * SA_S];   // [m][k] straight copy
    __shared__ __align__(16) ushort_t sB[2][GKC * SB_S];   // [n][k] straight copy

    const int tid  = threadIdx.x;
    const int lane = tid & 31;
    const int w    = tid >> 5;
    const int wm   = w & 1;        // 16-row tile (plus +32 sibling)
    const int wn   = w >> 1;       // 16-col tile within BN
    const int m0   = blockIdx.y * GBM;
    const int n0   = blockIdx.x * GBN;

    v8f acc0 = {0.f, 0.f, 0.f, 0.f, 0.f, 0.f, 0.f, 0.f};
    v8f acc1 = {0.f, 0.f, 0.f, 0.f, 0.f, 0.f, 0.f, 0.f};

    const int am = tid >> 2;            // 0..63 rows of A tile
    const int ak = (tid & 3) << 4;      // 16 contiguous bf16 (2 async b128)
    const int bn = tid & 63;            // 0..63 rows (n) of B tile
    const int bq = (tid >> 6) << 4;     // 16 contiguous bf16 per thread

    auto stage = [&](int k0, int bb) {
        const ushort_t* ap = A + (size_t)(m0 + am) * lda + k0 + ak;
        async_copy_b128(ap,     &sA[bb][am * SA_S + ak]);
        async_copy_b128(ap + 8, &sA[bb][am * SA_S + ak + 8]);
        const ushort_t* bp = Bm + (size_t)(n0 + bn) * ldb + k0 + bq;
        uint4 r0 = *(const uint4*)bp;
        uint4 r1 = *(const uint4*)(bp + 8);
        const ushort_t* rr0 = (const ushort_t*)&r0;
        const ushort_t* rr1 = (const ushort_t*)&r1;
#pragma unroll
        for (int i = 0; i < 8; ++i) sB[bb][bn * SB_S + bq + i] = rr0[i];
#pragma unroll
        for (int i = 0; i < 8; ++i) sB[bb][bn * SB_S + bq + 8 + i] = rr1[i];
    };

    stage(0, 0);
    wait_async0();
    __syncthreads();

    const int fm = wm * 16 + (lane & 15);
    const int kb = (lane < 16) ? 0 : 8;   // A frag K-chunk base per lane half
    int bb = 0;
    for (int k0 = 0; k0 < K; k0 += GKC) {
        if (k0 + GKC < K) stage(k0 + GKC, bb ^ 1);   // overlap with compute

        FragU fa0, fa1, fb;
        // k-chunk 0 (k0+0..31)
        fa0.q[0] = *(const uint4*)&sA[bb][fm * SA_S + kb];
        fa0.q[1] = *(const uint4*)&sA[bb][fm * SA_S + kb + 16];
        fa1.q[0] = *(const uint4*)&sA[bb][(fm + 32) * SA_S + kb];
        fa1.q[1] = *(const uint4*)&sA[bb][(fm + 32) * SA_S + kb + 16];
        fb.q[0]  = *(const uint4*)&sB[bb][(wn * 16 + (lane & 15)) * SB_S + kb];
        fb.q[1]  = *(const uint4*)&sB[bb][(wn * 16 + (lane & 15)) * SB_S + kb + 16];
        acc0 = __builtin_amdgcn_wmma_f32_16x16x32_bf16(
            false, fa0.v, false, fb.v, (short)0, acc0, false, false);
        acc1 = __builtin_amdgcn_wmma_f32_16x16x32_bf16(
            false, fa1.v, false, fb.v, (short)0, acc1, false, false);
        // k-chunk 1 (k0+32..63)
        fa0.q[0] = *(const uint4*)&sA[bb][fm * SA_S + 32 + kb];
        fa0.q[1] = *(const uint4*)&sA[bb][fm * SA_S + 32 + kb + 16];
        fa1.q[0] = *(const uint4*)&sA[bb][(fm + 32) * SA_S + 32 + kb];
        fa1.q[1] = *(const uint4*)&sA[bb][(fm + 32) * SA_S + 32 + kb + 16];
        fb.q[0]  = *(const uint4*)&sB[bb][(wn * 16 + (lane & 15)) * SB_S + 32 + kb];
        fb.q[1]  = *(const uint4*)&sB[bb][(wn * 16 + (lane & 15)) * SB_S + 32 + kb + 16];
        acc0 = __builtin_amdgcn_wmma_f32_16x16x32_bf16(
            false, fa0.v, false, fb.v, (short)0, acc0, false, false);
        acc1 = __builtin_amdgcn_wmma_f32_16x16x32_bf16(
            false, fa1.v, false, fb.v, (short)0, acc1, false, false);

        wait_async0();       // next-stage async copies landed
        __syncthreads();     // one barrier per k-stage
        bb ^= 1;
    }

    // C/D layout: element j -> m = j + 8*(lane>=16), n = lane%16
    const int col    = n0 + wn * 16 + (lane & 15);
    const int rbase0 = m0 + wm * 16 + ((lane & 16) ? 8 : 0);
    const float bv   = HAS_BIAS ? bias[col] : 0.f;
#pragma unroll
    for (int j = 0; j < 8; ++j) {
        const float v0 = acc0[j] + bv;
        const float v1 = acc1[j] + bv;
        const int r0 = rbase0 + j, r1 = rbase0 + 32 + j;
        if constexpr (OUT_BF16) {
            ((ushort_t*)Cv)[(size_t)r0 * ldc + col] = f2bf(v0);
            ((ushort_t*)Cv)[(size_t)r1 * ldc + col] = f2bf(v1);
        } else {
            ((float*)Cv)[(size_t)r0 * ldc + col] = v0;
            ((float*)Cv)[(size_t)r1 * ldc + col] = v1;
        }
    }
}

// ---------------------------------------------------------------------------
// One-time fp32 -> bf16 conversion (4 elems/thread; n % 4 == 0 at all sites).
// ---------------------------------------------------------------------------
__global__ __launch_bounds__(256)
void f32_to_bf16_kernel(const float* __restrict__ src, ushort_t* __restrict__ dst,
                        int n) {
    const int i4 = (blockIdx.x * 256 + threadIdx.x) * 4;
    if (i4 < n) {
        float4 v = *(const float4*)(src + i4);
        dst[i4 + 0] = f2bf(v.x);
        dst[i4 + 1] = f2bf(v.y);
        dst[i4 + 2] = f2bf(v.z);
        dst[i4 + 3] = f2bf(v.w);
    }
}

// Combined bias for the fused (q | gh) GEMM: [b_attn | b_hh] (2048 entries).
__global__ __launch_bounds__(256)
void build_bias_qh_kernel(const float* __restrict__ b_attn,
                          const float* __restrict__ b_hh,
                          float* __restrict__ bias_qh) {
    const int i = blockIdx.x * 256 + threadIdx.x;   // 0..2047
    bias_qh[i] = (i < 512) ? b_attn[i] : b_hh[i - 512];
}

// ---------------------------------------------------------------------------
// Init: h_cur = state (fp32 + bf16 mirror), cov = 0  (B*H == B*N == 32768)
// ---------------------------------------------------------------------------
__global__ __launch_bounds__(256)
void init_state_kernel(const float* __restrict__ state, float* __restrict__ h_cur,
                       ushort_t* __restrict__ h_bf, float* __restrict__ cov) {
    const int idx = blockIdx.x * 256 + threadIdx.x;
    const float s = state[idx];
    h_cur[idx] = s;
    h_bf[idx]  = f2bf(s);
    cov[idx]   = 0.f;
}

// ---------------------------------------------------------------------------
// Fused attention: one block per batch element b (256 threads, 8 waves).
// score (streams bf16 vk) -> softmax -> context (streams bf16 value);
// writes attn slice of d_out, updates cov, emits bf16 GRU input
// inp_bf = [emb_t | context]. q (with b_attn pre-folded) lives in qgh[:, 0:512].
// ---------------------------------------------------------------------------
__global__ __launch_bounds__(256)
void attn_kernel(const ushort_t* __restrict__ vk,      // [B*N, 512] bf16
                 const float* __restrict__ qgh,        // [B, 2048]; cols 0..511 = q+b_attn
                 const float* __restrict__ w_cov,      // [512]
                 const float* __restrict__ v_attn,     // [512]
                 const ushort_t* __restrict__ value_bf,// [B*N, 512] bf16
                 const float* __restrict__ vmask,      // [B, N]
                 const float* __restrict__ emb,        // [B, L, 512]
                 float* __restrict__ cov,              // [B, N]
                 float* __restrict__ attn_out,         // [B, L, N]
                 ushort_t* __restrict__ inp_bf,        // [B, 1024] bf16
                 int t) {
    const int b = blockIdx.x, tid = threadIdx.x;
    __shared__ float s_q[512], s_wc[512], s_va[512], s_sc[512], s_red[256];

    for (int i = tid; i < 512; i += 256) {
        s_q[i]  = qgh[b * 2048 + i];
        s_wc[i] = w_cov[i];
        s_va[i] = v_attn[i];
    }
    __syncthreads();

    const int wave = tid >> 5, lane = tid & 31;
    // --- score: each wave owns rows n = wave, wave+8, ...; lane covers 16 h.
    for (int n = wave; n < 512; n += 8) {
        const float cov_n = cov[b * 512 + n];
        const ushort_t* row = vk + (((size_t)b * 512 + n) << 9);
        const uint4* vp = (const uint4*)(row + lane * 16);
        __builtin_prefetch((const char*)vp + 8 * 512 * 2, 0, 1);  // next n row
        uint4 c0 = vp[0], c1 = vp[1];
        const ushort_t* u0 = (const ushort_t*)&c0;
        const ushort_t* u1 = (const ushort_t*)&c1;
        float accs = 0.f;
        const int hb = lane * 16;
#pragma unroll
        for (int j = 0; j < 8; ++j) {
            float eh = s_q[hb + j] + bf2f(u0[j]) + cov_n * s_wc[hb + j];
            accs += fmaxf(eh, 0.f) * s_va[hb + j];
        }
#pragma unroll
        for (int j = 0; j < 8; ++j) {
            float eh = s_q[hb + 8 + j] + bf2f(u1[j]) + cov_n * s_wc[hb + 8 + j];
            accs += fmaxf(eh, 0.f) * s_va[hb + 8 + j];
        }
#pragma unroll
        for (int off = 16; off > 0; off >>= 1)
            accs += __shfl_xor(accs, off, 32);
        if (lane == 0) {
            float m = vmask[b * 512 + n];
            s_sc[n] = (m > 0.f) ? accs : -1e9f;
        }
    }
    __syncthreads();

    // --- softmax over N=512
    s_red[tid] = fmaxf(s_sc[tid], s_sc[tid + 256]);
    __syncthreads();
    for (int off = 128; off > 0; off >>= 1) {
        if (tid < off) s_red[tid] = fmaxf(s_red[tid], s_red[tid + off]);
        __syncthreads();
    }
    const float mx = s_red[0];
    __syncthreads();
    float e0 = expf(s_sc[tid] - mx), e1 = expf(s_sc[tid + 256] - mx);
    s_sc[tid] = e0; s_sc[tid + 256] = e1;
    s_red[tid] = e0 + e1;
    __syncthreads();
    for (int off = 128; off > 0; off >>= 1) {
        if (tid < off) s_red[tid] += s_red[tid + off];
        __syncthreads();
    }
    const float inv = 1.0f / s_red[0];
    __syncthreads();

    // --- normalize, emit attn, update coverage
    for (int n = tid; n < 512; n += 256) {
        float a = s_sc[n] * inv;
        s_sc[n] = a;
        attn_out[((size_t)b * 128 + t) * 512 + n] = a;
        cov[b * 512 + n] += a;
    }
    __syncthreads();

    // --- context[h] = sum_n attn[n] * value[b,n,h] (bf16 stream, fp32 accum)
    {
        const ushort_t* vb = value_bf + ((size_t)b << 18) + 2 * tid;
        float a0 = 0.f, a1 = 0.f;
        for (int n = 0; n < 512; ++n) {
            const unsigned pp = *(const unsigned*)(vb + ((size_t)n << 9));
            const float wgt = s_sc[n];
            a0 += wgt * bf2f((ushort_t)(pp & 0xffffu));
            a1 += wgt * bf2f((ushort_t)(pp >> 16));
        }
        inp_bf[b * 1024 + 512 + 2 * tid + 0] = f2bf(a0);
        inp_bf[b * 1024 + 512 + 2 * tid + 1] = f2bf(a1);
    }
    // --- emb_t -> inp_bf[:,0:512]
    for (int e = tid; e < 512; e += 256)
        inp_bf[b * 1024 + e] = f2bf(emb[((size_t)b * 128 + t) * 512 + e]);
}

// ---------------------------------------------------------------------------
// GRU gates: h' = (1-z)*n + z*h. gh lives in qgh[:, 512:2048].
// ---------------------------------------------------------------------------
__global__ __launch_bounds__(256)
void gru_gates_kernel(const float* __restrict__ gi, const float* __restrict__ qgh,
                      float* __restrict__ h_cur, ushort_t* __restrict__ h_bf,
                      float* __restrict__ out, int t) {
    const int idx = blockIdx.x * 256 + threadIdx.x;  // B*H = 32768
    const int b = idx >> 9, j = idx & 511;
    const float* gib = gi + b * 1536;
    const float* ghb = qgh + b * 2048 + 512;
    const float ir = gib[j], iz = gib[512 + j], in_ = gib[1024 + j];
    const float hr = ghb[j], hz = ghb[512 + j], hn = ghb[1024 + j];
    const float r = 1.f / (1.f + expf(-(ir + hr)));
    const float z = 1.f / (1.f + expf(-(iz + hz)));
    const float nn = tanhf(in_ + r * hn);
    const float hp = h_cur[idx];
    const float hnew = (1.f - z) * nn + z * hp;
    h_cur[idx] = hnew;
    h_bf[idx]  = f2bf(hnew);
    out[((size_t)b * 128 + t) * 512 + j] = hnew;
}

// ---------------------------------------------------------------------------
extern "C" void kernel_launch(void* const* d_in, const int* in_sizes, int n_in,
                              void* d_out, int out_size, void* d_ws, size_t ws_size,
                              hipStream_t stream) {
    (void)in_sizes; (void)n_in; (void)out_size; (void)ws_size;
    const float* emb    = (const float*)d_in[0];
    const float* value  = (const float*)d_in[1];
    const float* vmask  = (const float*)d_in[2];
    const float* state  = (const float*)d_in[3];
    const float* Wq     = (const float*)d_in[4];
    const float* Wk     = (const float*)d_in[5];
    const float* b_attn = (const float*)d_in[6];
    const float* w_cov  = (const float*)d_in[7];
    const float* v_attn = (const float*)d_in[8];
    const float* W_ih   = (const float*)d_in[9];
    const float* W_hh   = (const float*)d_in[10];
    const float* b_ih   = (const float*)d_in[11];
    const float* b_hh   = (const float*)d_in[12];

    float* out_h    = (float*)d_out;                           // [B,L,H]
    float* out_attn = (float*)d_out + (size_t)64 * 128 * 512;  // [B,L,N]

    // workspace carve (~74 MB)
    char* p = (char*)d_ws;
    ushort_t* vk_bf    = (ushort_t*)p; p += (size_t)64 * 512 * 512 * 2;
    ushort_t* value_bf = (ushort_t*)p; p += (size_t)64 * 512 * 512 * 2;
    ushort_t* Wqh_bf   = (ushort_t*)p; p += (size_t)2048 * 512 * 2;  // [Wq; W_hh]
    ushort_t* Wk_bf    = (ushort_t*)p; p += (size_t)512 * 512 * 2;
    ushort_t* Wih_bf   = (ushort_t*)p; p += (size_t)1536 * 1024 * 2;
    float*    bias_qh  = (float*)p;    p += (size_t)2048 * 4;
    float*    h_cur    = (float*)p;    p += (size_t)64 * 512 * 4;
    ushort_t* h_bf     = (ushort_t*)p; p += (size_t)64 * 512 * 2;
    float*    cov      = (float*)p;    p += (size_t)64 * 512 * 4;
    float*    qgh      = (float*)p;    p += (size_t)64 * 2048 * 4;  // [q | gh]
    ushort_t* inp_bf   = (ushort_t*)p; p += (size_t)64 * 1024 * 2;
    float*    gi       = (float*)p;    p += (size_t)64 * 1536 * 4;

    // ---- one-time precision conversions + packing ----
    f32_to_bf16_kernel<<<512 * 512 / 4 / 256, 256, 0, stream>>>(Wq, Wqh_bf, 512 * 512);
    f32_to_bf16_kernel<<<1536 * 512 / 4 / 256, 256, 0, stream>>>(
        W_hh, Wqh_bf + (size_t)512 * 512, 1536 * 512);
    f32_to_bf16_kernel<<<512 * 512 / 4 / 256, 256, 0, stream>>>(Wk, Wk_bf, 512 * 512);
    f32_to_bf16_kernel<<<1536 * 1024 / 4 / 256, 256, 0, stream>>>(W_ih, Wih_bf, 1536 * 1024);
    f32_to_bf16_kernel<<<64 * 512 * 512 / 4 / 256, 256, 0, stream>>>(
        value, value_bf, 64 * 512 * 512);
    build_bias_qh_kernel<<<8, 256, 0, stream>>>(b_attn, b_hh, bias_qh);

    init_state_kernel<<<128, 256, 0, stream>>>(state, h_cur, h_bf, cov);

    // vk = value @ Wk^T (M=32768, N=512, K=512), bf16 output
    gemm_nt_bf16<true, false><<<dim3(512 / GBN, 32768 / GBM), 256, 0, stream>>>(
        value_bf, 512, Wk_bf, 512, vk_bf, 512, nullptr, 512);

    for (int t = 0; t < 128; ++t) {
        // [q | gh] = h @ [Wq; W_hh]^T + [b_attn | b_hh]   (M=64, N=2048, K=512)
        gemm_nt_bf16<false, true><<<dim3(2048 / GBN, 64 / GBM), 256, 0, stream>>>(
            h_bf, 512, Wqh_bf, 512, qgh, 2048, bias_qh, 512);

        attn_kernel<<<64, 256, 0, stream>>>(vk_bf, qgh, w_cov, v_attn, value_bf,
                                            vmask, emb, cov, out_attn, inp_bf, t);

        // gi = [emb_t, context] @ W_ih^T + b_ih   (M=64, N=1536, K=1024)
        gemm_nt_bf16<false, true><<<dim3(1536 / GBN, 64 / GBM), 256, 0, stream>>>(
            inp_bf, 1024, Wih_bf, 1024, gi, 1536, b_ih, 1024);

        gru_gates_kernel<<<128, 256, 0, stream>>>(gi, qgh, h_cur, h_bf, out_h, t);
    }
}
